// EmbeddingLoss_54709293416572
// MI455X (gfx1250) — matrix operand order repaired and checked by reading, not compiled, not run on hardware
//
#include <hip/hip_runtime.h>

// CDNA5 / gfx1250: wave32, v_wmma_f32_16x16x32_f16 + async global->LDS pipeline.
typedef _Float16 v8h  __attribute__((ext_vector_type(8)));
typedef _Float16 v16h __attribute__((ext_vector_type(16)));
typedef float    v8f  __attribute__((ext_vector_type(8)));

#define TH_SIM_MIN  0.9f
#define TH_DIFF_MAX 0.1f
#define BIGF        1e9f

// Low 32 bits of a generic pointer to __shared__ == LDS byte address
static __device__ __forceinline__ unsigned int ldsAddr(const void* p) {
    return (unsigned int)(uintptr_t)p;
}

// One wave = one batch element per pipeline stage.
//  stage (8KB f32, async-filled)  ->  E (4KB f16, normalized+transposed)
//  -> 16x v_wmma_f32_16x16x32_f16 -> fused masked min/max row reductions.
__global__ __launch_bounds__(128) void emb_loss_main(
    const float* __restrict__ emb, const int* __restrict__ lbl,
    float* __restrict__ partial, int B)
{
    // per wave: 8KB f32 staging + 4KB f16 tile; 4 waves -> 48KB
    __shared__ __align__(16) unsigned char sLds[4 * (8192 + 4096)];
    __shared__ float sRed[4];

    const int tid   = threadIdx.x;
    const int lane  = tid & 31;
    const int wid   = tid >> 5;
    const int wgl   = blockIdx.x * 4 + wid;      // global wave id
    const int wstr  = gridDim.x * 4;             // wave stride over batches

    float*     stage = (float*)(sLds + wid * 12288);
    _Float16*  E     = (_Float16*)(sLds + wid * 12288 + 8192);
    const unsigned int stageA = ldsAddr(stage);

    // ---- async tile fetch: 8KB = 16 issues x (32 lanes x 16B) ----
    auto issueTile = [&](int bb) {
        unsigned long long g =
            (unsigned long long)(uintptr_t)(emb + (size_t)bb * 2048) +
            (unsigned long long)(lane * 16u);
        unsigned int l = stageA + lane * 16u;
        #pragma unroll
        for (int k = 0; k < 16; ++k) {
            asm volatile("global_load_async_to_lds_b128 %0, %1, off"
                         :: "v"(l), "v"(g) : "memory");
            l += 512u;
            g += 512ull;
        }
    };

    const int colIdx = lane & 15;
    const int hiHalf = (lane >= 16);
    const int aSel   = hiHalf ? 8  : 0;   // A frag: K{0-7,16-23} vs {8-15,24-31}
    const int bSel   = hiHalf ? 8  : 0;   // B frag: K{0-7,16-23} vs {8-15,24-31}
    const int rbase  = hiHalf ? 8  : 0;   // C/D rows r vs r+8
    const int t0     = lane * 2;          // columns owned in normalize phase

    float runAcc = 0.0f;

    if (wgl < B) issueTile(wgl);                      // pipeline prolog

    for (int b = wgl; b < B; b += wstr) {
        // wait for async fill of the staging buffer
        asm volatile("s_wait_asynccnt 0" ::: "memory");

        // ---- labels -> 64-bit class mask (bit t set => label==1 "real") ----
        const int l0 = lbl[(size_t)b * 64 + lane];
        const int l1 = lbl[(size_t)b * 64 + 32 + lane];
        unsigned long long mlo = __ballot(l0 != 0);
        unsigned long long mhi = __ballot(l1 != 0);
        const unsigned long long mask = (mlo & 0xffffffffull) | (mhi << 32);
        const int n_real = __popcll(mask);
        const int n_fake = 64 - n_real;
        const float invR = 1.0f / (float)(n_real > 1 ? n_real : 1);
        const float invF = 1.0f / (float)(n_fake > 1 ? n_fake : 1);
        const float has_both = (n_real > 0 && n_fake > 0) ? 1.0f : 0.0f;

        // ---- normalize: stage[d][t] f32 -> E[t][d] f16 ----
        {
            float vx[32], vy[32];
            float ssx = 0.f, ssy = 0.f;
            #pragma unroll
            for (int d = 0; d < 32; ++d) {
                const float2 p = *(const float2*)(stage + d * 64 + t0);
                vx[d] = p.x; vy[d] = p.y;
                ssx += p.x * p.x; ssy += p.y * p.y;
            }
            const float sx = __frsqrt_rn(fmaxf(ssx, 1e-8f));
            const float sy = __frsqrt_rn(fmaxf(ssy, 1e-8f));
            #pragma unroll
            for (int q = 0; q < 4; ++q) {
                v8h hx, hy;
                #pragma unroll
                for (int k = 0; k < 8; ++k) {
                    hx[k] = (_Float16)(vx[q * 8 + k] * sx);
                    hy[k] = (_Float16)(vy[q * 8 + k] * sy);
                }
                *(v8h*)(E + t0 * 32 + q * 8)       = hx;
                *(v8h*)(E + (t0 + 1) * 32 + q * 8) = hy;
            }
        }

        // staging reads retired -> safe to refill for the next batch; the
        // async copy then overlaps with the WMMA/reduction work below.
        asm volatile("s_wait_dscnt 0" ::: "memory");
        if (b + wstr < B) issueTile(b + wstr);

        // ---- hoisted B fragments (reused by all 4 tile-rows) ----
        v16h Bf[4];
        #pragma unroll
        for (int j = 0; j < 4; ++j) {
            const int colB = j * 16 + colIdx;
            const v8h blo = *(const v8h*)(E + colB * 32 + bSel);
            const v8h bhi = *(const v8h*)(E + colB * 32 + 16 + bSel);
            Bf[j] = __builtin_shufflevector(blo, bhi,
                0, 1, 2, 3, 4, 5, 6, 7, 8, 9, 10, 11, 12, 13, 14, 15);
        }

        float acc = 0.0f;
        #pragma unroll
        for (int i = 0; i < 4; ++i) {
            const int colA = i * 16 + colIdx;
            const v8h alo = *(const v8h*)(E + colA * 32 + aSel);
            const v8h ahi = *(const v8h*)(E + colA * 32 + 16 + aSel);
            const v16h A = __builtin_shufflevector(alo, ahi,
                0, 1, 2, 3, 4, 5, 6, 7, 8, 9, 10, 11, 12, 13, 14, 15);

            int rowBit[8];
            float minS[8], maxC[8];
            #pragma unroll
            for (int r = 0; r < 8; ++r) {
                rowBit[r] = (int)((mask >> (i * 16 + rbase + r)) & 1ull);
                minS[r] = BIGF; maxC[r] = -BIGF;
            }

            #pragma unroll
            for (int j = 0; j < 4; ++j) {
                v8f c = {};
                const v8f d = __builtin_amdgcn_wmma_f32_16x16x32_f16(
                    false, A, false, Bf[j], (short)0, c, false, false);

                const int colBit = (int)((mask >> (j * 16 + colIdx)) & 1ull);
                #pragma unroll
                for (int r = 0; r < 8; ++r) {
                    const float v = d[r];
                    if (colBit == rowBit[r]) minS[r] = fminf(minS[r], v);
                    else                     maxC[r] = fmaxf(maxC[r], v);
                }
            }

            // reduce across the 16-lane half (xor 1..8 stays within a half)
            #pragma unroll
            for (int r = 0; r < 8; ++r) {
                float mn = minS[r], mx = maxC[r];
                #pragma unroll
                for (int off = 1; off <= 8; off <<= 1) {
                    mn = fminf(mn, __shfl_xor(mn, off, 32));
                    mx = fmaxf(mx, __shfl_xor(mx, off, 32));
                }
                const float lrow = fmaxf(TH_SIM_MIN - mn, 0.0f)
                                 + fmaxf(mx - TH_DIFF_MAX, 0.0f);
                acc += lrow * (rowBit[r] ? invR : invF);
            }
        }
        runAcc += acc * has_both;
    }

    // one wave-wide sum at the end; each row value replicated on 16 lanes
    #pragma unroll
    for (int off = 1; off <= 16; off <<= 1)
        runAcc += __shfl_xor(runAcc, off, 32);

    if (lane == 0) sRed[wid] = runAcc * (1.0f / 16.0f);
    __syncthreads();
    if (tid == 0)
        partial[blockIdx.x] = sRed[0] + sRed[1] + sRed[2] + sRed[3];
}

// Deterministic fixed-tree final reduction (no float atomics).
__global__ __launch_bounds__(256) void emb_loss_reduce(
    const float* __restrict__ partial, int n, float* __restrict__ out, float invB)
{
    __shared__ float s[256];
    float a = 0.0f;
    for (int i = threadIdx.x; i < n; i += 256) a += partial[i];
    s[threadIdx.x] = a;
    __syncthreads();
    for (int w = 128; w > 0; w >>= 1) {
        if ((int)threadIdx.x < w) s[threadIdx.x] += s[threadIdx.x + w];
        __syncthreads();
    }
    if (threadIdx.x == 0) out[0] = s[0] * invB;
}

extern "C" void kernel_launch(void* const* d_in, const int* in_sizes, int n_in,
                              void* d_out, int out_size, void* d_ws, size_t ws_size,
                              hipStream_t stream) {
    const float* emb = (const float*)d_in[0];
    const int*   lbl = (const int*)d_in[1];
    float*       out = (float*)d_out;
    float*       ws  = (float*)d_ws;

    const int B      = in_sizes[1] / 64;          // label is [B, 64]
    const int needed = (B + 3) / 4;               // blocks if 1 batch per wave
    const int nblk   = needed < 2048 ? needed : 2048;  // ~4 batches per wave

    emb_loss_main<<<nblk, 128, 0, stream>>>(emb, lbl, ws, B);
    emb_loss_reduce<<<1, 256, 0, stream>>>(ws, nblk, out, 1.0f / (float)B);
}
